// MambaLayer_47339129537009
// MI455X (gfx1250) — matrix operand order repaired and checked by reading, not compiled, not run on hardware
//
#include <hip/hip_runtime.h>
#include <stdint.h>

// ---------------- problem constants ----------------
#define T_LEN 2048
#define NBATCH 32
#define DIM 512
#define M_ROWS (T_LEN * NBATCH)   // 65536 rows (T*B)
#define N1 (2 * DIM)              // 1024 (in_proj out)
#define K_DIM DIM                 // 512
#define KCH 32                    // K-chunk per WMMA stage
#define NCHUNK (K_DIM / KCH)      // 16
#define LDA 40                    // padded LDS row stride (32 data + 8 pad bf16)

// ---------------- vector types ----------------
typedef __attribute__((ext_vector_type(16))) __bf16 v16bf;
typedef __attribute__((ext_vector_type(8)))  __bf16 v8bf;
typedef __attribute__((ext_vector_type(8)))  float  v8f;
typedef __attribute__((ext_vector_type(4)))  unsigned int v4u;
typedef __attribute__((ext_vector_type(8)))  int v8i;
typedef __attribute__((ext_vector_type(4)))  int v4i;

#if defined(__has_builtin)
#if __has_builtin(__builtin_amdgcn_tensor_load_to_lds) && \
    __has_builtin(__builtin_amdgcn_s_wait_tensorcnt)
#define USE_TDM 1
#endif
#endif
#ifndef USE_TDM
#define USE_TDM 0
#endif

static __device__ __forceinline__ unsigned short f2bf(float f) {
  // round-to-nearest-even fp32 -> bf16
  unsigned int u = __float_as_uint(f);
  u += 0x7FFFu + ((u >> 16) & 1u);
  return (unsigned short)(u >> 16);
}

static __device__ __forceinline__ v16bf cat16(v8bf lo, v8bf hi) {
  return __builtin_shufflevector(lo, hi, 0,1,2,3,4,5,6,7,8,9,10,11,12,13,14,15);
}

#if USE_TDM
typedef __attribute__((address_space(3))) void lds_void;
static __device__ __forceinline__ unsigned lds_addr_of(void* p) {
  // flat -> LDS addrspace cast truncates to the wave's LDS byte offset (ISA 10.2)
  return (unsigned)(uintptr_t)(lds_void*)p;
}

// TDM: DMA a 128-row x 32-elem bf16 tile (row stride = K_DIM elems) from global
// into LDS, with HW padding of 4 DWORDs after every 16 DWORDs -> LDS rows of
// LDA=40 bf16 elems (bank-conflict-free WMMA fragment loads).
static __device__ __forceinline__ void tdm_tile_128x32(
    unsigned lds_byte_addr, const unsigned short* gsrc, unsigned nrows) {
  unsigned long long ga = (unsigned long long)(uintptr_t)gsrc;
  v4u g0;
  g0[0] = 1u;                                   // count=1, user descriptor
  g0[1] = lds_byte_addr;                        // lds_addr
  g0[2] = (unsigned)ga;                         // global_addr[31:0]
  g0[3] = (unsigned)(ga >> 32) | (2u << 30);    // global_addr[56:32] | type=2
  v8i g1;
  g1[0] = (int)((1u << 16)     // data_size = 2 bytes
              | (1u << 20)     // pad_enable
              | (3u << 22)     // pad_interval code 3 = 16 DWORDs (64B data)
              | (3u << 25));   // pad_amount  code 3 = 4 DWORDs (16B pad)
  g1[1] = (int)((unsigned)K_DIM << 16);                        // tensor_dim0 lo
  g1[2] = (int)(((unsigned)K_DIM >> 16) | ((nrows & 0xffffu) << 16));
  g1[3] = (int)((nrows >> 16) | ((unsigned)KCH << 16));        // tile_dim0 = 32
  g1[4] = 128;                                                 // tile_dim1 = 128
  g1[5] = K_DIM;                                               // tensor_dim0_stride
  g1[6] = 0;
  g1[7] = 0;
  v4i gz; gz[0] = gz[1] = gz[2] = gz[3] = 0;                   // <=2D: groups 2,3 zero
#if __clang_major__ >= 23
  v8i gz8; gz8[0]=gz8[1]=gz8[2]=gz8[3]=gz8[4]=gz8[5]=gz8[6]=gz8[7]=0;
  __builtin_amdgcn_tensor_load_to_lds(g0, g1, gz, gz, gz8, 0);
#else
  __builtin_amdgcn_tensor_load_to_lds(g0, g1, gz, gz, 0);
#endif
}
#endif  // USE_TDM

// ---------------- fp32 -> bf16 convert ----------------
__global__ __launch_bounds__(256) void cvt_bf16_kernel(const float* __restrict__ src,
                                                       unsigned short* __restrict__ dst,
                                                       int n4) {
  int i = blockIdx.x * 256 + threadIdx.x;
  if (i < n4) {
    float4 f = ((const float4*)src)[i];
    uint2 o;
    o.x = (unsigned int)f2bf(f.x) | ((unsigned int)f2bf(f.y) << 16);
    o.y = (unsigned int)f2bf(f.z) | ((unsigned int)f2bf(f.w) << 16);
    ((uint2*)dst)[i] = o;
  }
}

// ---------------- shared WMMA mainloop ----------------
// block tile 128x128 (8 waves, wave tile 64x32 = 4x2 WMMA 16x16 tiles), K-step 32.
// A: [M,K] bf16 row-major; W: [NROWS_W,K] bf16 row-major (K-contiguous per output
// column == WMMA B-fragment per-lane layout, no transpose needed).
template <int NROWS_W>
static __device__ __forceinline__ void wmma_mainloop(
    const unsigned short* __restrict__ aptr, const unsigned short* __restrict__ wptr,
    int m0, int n0, int wm, int wn, int lane, int wid,
    __bf16 (*As)[128 * LDA], __bf16 (*Ws)[128 * LDA], v8f (*acc)[2]) {
  const int l15 = lane & 15;
  const int lhalf = lane >> 4;

#if USE_TDM
  if (wid == 0) {
    tdm_tile_128x32(lds_addr_of(&As[0][0]), aptr + (size_t)m0 * K_DIM, (unsigned)M_ROWS);
    tdm_tile_128x32(lds_addr_of(&Ws[0][0]), wptr + (size_t)n0 * K_DIM, (unsigned)NROWS_W);
  }
#endif

  for (int c = 0; c < NCHUNK; ++c) {
    const int cur = c & 1;
#if USE_TDM
    if (wid == 0) {
      if (c + 1 < NCHUNK) {  // prefetch next K-chunk into the other buffer
        const int k0n = (c + 1) * KCH;
        tdm_tile_128x32(lds_addr_of(&As[cur ^ 1][0]),
                        aptr + (size_t)m0 * K_DIM + k0n, (unsigned)M_ROWS);
        tdm_tile_128x32(lds_addr_of(&Ws[cur ^ 1][0]),
                        wptr + (size_t)n0 * K_DIM + k0n, (unsigned)NROWS_W);
        __builtin_amdgcn_s_wait_tensorcnt(2);  // current stage's 2 DMAs complete
      } else {
        __builtin_amdgcn_s_wait_tensorcnt(0);
      }
    }
    __syncthreads();  // publish TDM-written LDS to all waves
#else
    {  // vmem staging fallback: straight-line, 2x 16B chunks per thread per tile
      const int k0 = c * KCH;
      const int tid = wid * 32 + lane;
#pragma unroll
      for (int it = 0; it < 2; ++it) {
        int i = tid + it * 256;
        int row = i >> 2, cc = i & 3;
        *(uint4*)(&As[cur][row * LDA + cc * 8]) =
            *(const uint4*)(aptr + (size_t)(m0 + row) * K_DIM + k0 + cc * 8);
        *(uint4*)(&Ws[cur][row * LDA + cc * 8]) =
            *(const uint4*)(wptr + (size_t)(n0 + row) * K_DIM + k0 + cc * 8);
      }
    }
    __syncthreads();
#endif

    // A fragment (ISA 16-bit A 16x32): lane M=base+l15; elems 0..7 = K c..c+7,
    // elems 8..15 = K 16+c..16+c+7, c = lhalf*8. Two 16B LDS loads each.
    v16bf afrag[4];
#pragma unroll
    for (int mt = 0; mt < 4; ++mt) {
      int base = (wm + mt * 16 + l15) * LDA + lhalf * 8;
      afrag[mt] = cat16(*(const v8bf*)(&As[cur][base]),
                        *(const v8bf*)(&As[cur][base + 16]));
    }
    // B fragment: lane N=base+l15; elems 0..7 = K c..c+7, elems 8..15 = K 16+c..16+c+7, c = lhalf*8.
    v16bf bfrag[2];
#pragma unroll
    for (int nt = 0; nt < 2; ++nt) {
      int base = (wn + nt * 16 + l15) * LDA + lhalf * 8;
      bfrag[nt] = cat16(*(const v8bf*)(&Ws[cur][base]),
                        *(const v8bf*)(&Ws[cur][base + 16]));
    }
#pragma unroll
    for (int mt = 0; mt < 4; ++mt)
#pragma unroll
      for (int nt = 0; nt < 2; ++nt)
        acc[mt][nt] = __builtin_amdgcn_wmma_f32_16x16x32_bf16(
            false, afrag[mt], false, bfrag[nt], (short)0, acc[mt][nt], false, false);
    __syncthreads();  // all waves done reading buf[cur] before it is refilled
  }
}

// ---------------- GEMM1: zg = x @ W_in^T, fused gating epilogue ----------------
__global__ __launch_bounds__(256) void gemm1_gate_kernel(
    const unsigned short* __restrict__ xb, const unsigned short* __restrict__ wib,
    const float* __restrict__ b_in, const float* __restrict__ Avec,
    const float* __restrict__ Bvec, float* __restrict__ bz, float* __restrict__ ga) {
  __shared__ __align__(16) __bf16 As[2][128 * LDA];
  __shared__ __align__(16) __bf16 Ws[2][128 * LDA];

  const int tid = threadIdx.x;
  const int lane = tid & 31, wid = tid >> 5;
  const int l15 = lane & 15, lhalf = lane >> 4;
  const int m0 = blockIdx.x * 128, n0 = blockIdx.y * 128;
  const int wm = (wid & 1) * 64, wn = (wid >> 1) * 32;

  const v8f vzero = {0.f, 0.f, 0.f, 0.f, 0.f, 0.f, 0.f, 0.f};
  v8f acc[4][2];
#pragma unroll
  for (int i = 0; i < 4; ++i)
#pragma unroll
    for (int j = 0; j < 2; ++j) acc[i][j] = vzero;

  wmma_mainloop<N1>(xb, wib, m0, n0, wm, wn, lane, wid, As, Ws, acc);

  // epilogue: z-half -> Bz = B*(zg+b); g-half -> gA = sigmoid(zg+b)*A
#pragma unroll
  for (int mt = 0; mt < 4; ++mt) {
#pragma unroll
    for (int nt = 0; nt < 2; ++nt) {
      int gmb = m0 + wm + mt * 16 + lhalf * 8;  // + r below (C/D VGPR layout)
      int gn = n0 + wn + nt * 16 + l15;
      float bias = b_in[gn];
      if (gn < DIM) {
        float bc = Bvec[gn];
#pragma unroll
        for (int r = 0; r < 8; ++r)
          bz[(size_t)(gmb + r) * DIM + gn] = bc * (acc[mt][nt][r] + bias);
      } else {
        int d = gn - DIM;
        float ac = Avec[d];
#pragma unroll
        for (int r = 0; r < 8; ++r) {
          float g = 1.f / (1.f + __expf(-(acc[mt][nt][r] + bias)));
          ga[(size_t)(gmb + r) * DIM + d] = g * ac;
        }
      }
    }
  }
}

// ---------------- diagonal recurrence h_t = gA_t*h_{t-1} + Bz_t ----------------
__global__ __launch_bounds__(256) void scan_kernel(const float* __restrict__ ga,
                                                   const float* __restrict__ bz,
                                                   unsigned short* __restrict__ hb) {
  const int tid = blockIdx.x * 256 + threadIdx.x;  // 0 .. B*D-1 (coalesced over D)
  const size_t BD = (size_t)NBATCH * DIM;
  float h = 0.f;
  size_t off = tid;
#pragma unroll 8
  for (int t = 0; t < T_LEN; ++t) {
    h = __builtin_fmaf(ga[off], h, bz[off]);
    hb[off] = f2bf(h);
    off += BD;
  }
}

// ---------------- GEMM2: out = h @ W_out^T + b_out + x (residual) ----------------
__global__ __launch_bounds__(256) void gemm2_resid_kernel(
    const unsigned short* __restrict__ hbuf, const unsigned short* __restrict__ wob,
    const float* __restrict__ b_out, const float* __restrict__ x,
    float* __restrict__ out) {
  __shared__ __align__(16) __bf16 As[2][128 * LDA];
  __shared__ __align__(16) __bf16 Ws[2][128 * LDA];

  const int tid = threadIdx.x;
  const int lane = tid & 31, wid = tid >> 5;
  const int l15 = lane & 15, lhalf = lane >> 4;
  const int m0 = blockIdx.x * 128, n0 = blockIdx.y * 128;
  const int wm = (wid & 1) * 64, wn = (wid >> 1) * 32;

  const v8f vzero = {0.f, 0.f, 0.f, 0.f, 0.f, 0.f, 0.f, 0.f};
  v8f acc[4][2];
#pragma unroll
  for (int i = 0; i < 4; ++i)
#pragma unroll
    for (int j = 0; j < 2; ++j) acc[i][j] = vzero;

  wmma_mainloop<DIM>(hbuf, wob, m0, n0, wm, wn, lane, wid, As, Ws, acc);

#pragma unroll
  for (int mt = 0; mt < 4; ++mt) {
#pragma unroll
    for (int nt = 0; nt < 2; ++nt) {
      int gmb = m0 + wm + mt * 16 + lhalf * 8;
      int gn = n0 + wn + nt * 16 + l15;
      float bias = b_out[gn];
#pragma unroll
      for (int r = 0; r < 8; ++r) {
        size_t idx = (size_t)(gmb + r) * DIM + gn;
        out[idx] = acc[mt][nt][r] + bias + x[idx];
      }
    }
  }
}

// ---------------- in-place LayerNorm over last dim (512) ----------------
__global__ __launch_bounds__(256) void ln_kernel(float* __restrict__ y,
                                                 const float* __restrict__ gamma,
                                                 const float* __restrict__ beta) {
  __shared__ float red[256];
  const size_t row = blockIdx.x;
  float* p = y + row * DIM;
  const int t = threadIdx.x;
  float v0 = p[t], v1 = p[t + 256];

  red[t] = v0 + v1;
  __syncthreads();
#pragma unroll
  for (int o = 128; o > 0; o >>= 1) {
    if (t < o) red[t] += red[t + o];
    __syncthreads();
  }
  float mu = red[0] * (1.f / (float)DIM);
  __syncthreads();

  float d0 = v0 - mu, d1 = v1 - mu;
  red[t] = d0 * d0 + d1 * d1;
  __syncthreads();
#pragma unroll
  for (int o = 128; o > 0; o >>= 1) {
    if (t < o) red[t] += red[t + o];
    __syncthreads();
  }
  float rs = rsqrtf(red[0] * (1.f / (float)DIM) + 1e-5f);

  p[t]       = d0 * rs * gamma[t] + beta[t];
  p[t + 256] = d1 * rs * gamma[t + 256] + beta[t + 256];
}

// ---------------- launch ----------------
extern "C" void kernel_launch(void* const* d_in, const int* in_sizes, int n_in,
                              void* d_out, int out_size, void* d_ws, size_t ws_size,
                              hipStream_t stream) {
  (void)in_sizes; (void)n_in; (void)out_size; (void)ws_size;
  const float* x     = (const float*)d_in[0];
  const float* W_in  = (const float*)d_in[1];
  const float* b_in  = (const float*)d_in[2];
  const float* W_out = (const float*)d_in[3];
  const float* b_out = (const float*)d_in[4];
  const float* Avec  = (const float*)d_in[5];
  const float* Bvec  = (const float*)d_in[6];
  const float* gamma = (const float*)d_in[7];
  const float* beta  = (const float*)d_in[8];
  float* out = (float*)d_out;

  char* ws = (char*)d_ws;
  unsigned short* xb  = (unsigned short*)ws; ws += (size_t)M_ROWS * DIM * 2;  // x bf16
  unsigned short* wib = (unsigned short*)ws; ws += (size_t)N1 * DIM * 2;      // W_in bf16
  unsigned short* wob = (unsigned short*)ws; ws += (size_t)DIM * DIM * 2;     // W_out bf16
  float* bz = (float*)ws;                    ws += (size_t)M_ROWS * DIM * 4;  // B*z
  float* ga = (float*)ws;                    ws += (size_t)M_ROWS * DIM * 4;  // g*A
  unsigned short* hb = (unsigned short*)ws;                                   // h bf16

  // fp32 -> bf16
  {
    int n4 = (M_ROWS * DIM) / 4;
    cvt_bf16_kernel<<<(n4 + 255) / 256, 256, 0, stream>>>(x, xb, n4);
    n4 = (N1 * DIM) / 4;
    cvt_bf16_kernel<<<(n4 + 255) / 256, 256, 0, stream>>>(W_in, wib, n4);
    n4 = (DIM * DIM) / 4;
    cvt_bf16_kernel<<<(n4 + 255) / 256, 256, 0, stream>>>(W_out, wob, n4);
  }

  // GEMM1 + gating: [65536,512] x [512,1024]
  gemm1_gate_kernel<<<dim3(M_ROWS / 128, N1 / 128), 256, 0, stream>>>(
      xb, wib, b_in, Avec, Bvec, bz, ga);

  // recurrence over T
  scan_kernel<<<(NBATCH * DIM) / 256, 256, 0, stream>>>(ga, bz, hb);

  // GEMM2 + residual: [65536,512] x [512,512]
  gemm2_resid_kernel<<<dim3(M_ROWS / 128, DIM / 128), 256, 0, stream>>>(
      hb, wob, b_out, x, out);

  // LayerNorm in place on d_out
  ln_kernel<<<M_ROWS, 256, 0, stream>>>(out, gamma, beta);
}
